// SPSAGE_26388279067153
// MI455X (gfx1250) — matrix operand order repaired and checked by reading, not compiled, not run on hardware
//
#include <hip/hip_runtime.h>

typedef __attribute__((ext_vector_type(16))) _Float16 v16h;
typedef __attribute__((ext_vector_type(8)))  _Float16 v8h;
typedef __attribute__((ext_vector_type(8)))  float    v8f;

#define N0_  1100000
#define N1_  110000
#define N2_  11000
#define N3_  1024
#define FAN  10
#define FDIM 256
#define CLS_ 47

// ---------------------------------------------------------------------------
// Repack f32 [W_self;W_neigh] (each [256 x Nreal] row-major) into the WMMA
// B-fragment order this code uses for V_WMMA_F32_16X16X32_F16 (A matches it):
//   per (n-tile t, k-step s): 32 lanes x 16 halves, lane holds N = t*16 + (lane&15),
//   element e corresponds to K = s*32 + (lane>=16 ? 16 : 0) + e.
// Layer-2 columns 47..63 are zero padding.
// ---------------------------------------------------------------------------
__global__ void pack_w_kernel(const float* __restrict__ Wself,
                              const float* __restrict__ Wneigh,
                              int Nreal, int Ntiles,
                              _Float16* __restrict__ Wp)
{
    int idx = blockIdx.x * blockDim.x + threadIdx.x;
    int total = Ntiles * 16 * 32 * 16;
    if (idx >= total) return;
    int e    =  idx        & 15;
    int lane = (idx >> 4)  & 31;
    int s    = (idx >> 9)  & 15;
    int t    =  idx >> 13;
    int k = s * 32 + ((lane >= 16) ? 16 : 0) + e;   // 0..511
    int n = t * 16 + (lane & 15);
    float v = 0.0f;
    if (n < Nreal)
        v = (k < FDIM) ? Wself[(size_t)k * Nreal + n]
                       : Wneigh[(size_t)(k - FDIM) * Nreal + n];
    Wp[idx] = (_Float16)v;
}

// ---------------------------------------------------------------------------
// Neighbor mean: dst nodes have exactly FAN contiguous edges (dst = repeat(arange)).
// One block per dst node, 256 threads = one feature column each.
// ---------------------------------------------------------------------------
template <bool HF32>
__global__ void sage_agg_kernel(const void* __restrict__ hsrc,
                                const int*  __restrict__ src,
                                _Float16*   __restrict__ agg)
{
    int d   = blockIdx.x;
    int col = threadIdx.x;
    const int* sp = src + (size_t)d * FAN;
    float sum = 0.0f;
#pragma unroll
    for (int e = 0; e < FAN; ++e) {
        int sn = sp[e];                       // uniform per block -> scalar load
        if (HF32) sum += ((const float*)hsrc)[(size_t)sn * FDIM + col];
        else      sum += (float)(((const _Float16*)hsrc)[(size_t)sn * FDIM + col]);
    }
    agg[(size_t)d * FDIM + col] = (_Float16)(sum * 0.1f);   // cnt == 10 exactly
}

// ---------------------------------------------------------------------------
// Fused GEMM: out[m, n] = selfA[m, 0:256] . Wp_self + aggA[m, 0:256] . Wp_neigh + b
// Block = 128 threads (4 wave32 waves), covers 16 output rows x (Ntiles*16) cols.
// A tile (16 x 512 f16) staged in LDS (pitch 520 halves -> conflict-free b128
// fragment reads). Per-wave: TPW accumulators (n-tiles stride 4), 16 k-steps of
// v_wmma_f32_16x16x32_f16.
// ---------------------------------------------------------------------------
template <int TPW, bool SELFF32, bool OUTF16, bool RELU>
__global__ __launch_bounds__(128)
void sage_gemm_kernel(const void*      __restrict__ selfA,   // [M][256] f32 or f16
                      const _Float16*  __restrict__ aggA,    // [M][256] f16
                      const _Float16*  __restrict__ Wp,      // packed B fragments
                      const float*     __restrict__ bias,    // [NcolsReal]
                      int M, int NcolsReal, int outStride,
                      void* __restrict__ outp)
{
    __shared__ _Float16 As[16][520];          // 16 rows x 512 cols, pitch 520
    const int tid   = threadIdx.x;
    const int mbase = blockIdx.x * 16;

    // ---- stage A tile: cols 0..255 = self(h_dst), 256..511 = agg; pad rows = 0
    for (int base = tid * 4; base < 16 * 512; base += 128 * 4) {
        int r   = base >> 9;
        int col = base & 511;
        int gm  = mbase + r;
        _Float16 v0 = (_Float16)0.f, v1 = v0, v2 = v0, v3 = v0;
        if (gm < M) {
            if (col < FDIM) {
                if (SELFF32) {
                    const float* p = (const float*)selfA + (size_t)gm * FDIM + col;
                    v0 = (_Float16)p[0]; v1 = (_Float16)p[1];
                    v2 = (_Float16)p[2]; v3 = (_Float16)p[3];
                } else {
                    const _Float16* p = (const _Float16*)selfA + (size_t)gm * FDIM + col;
                    v0 = p[0]; v1 = p[1]; v2 = p[2]; v3 = p[3];
                }
            } else {
                const _Float16* p = aggA + (size_t)gm * FDIM + (col - FDIM);
                v0 = p[0]; v1 = p[1]; v2 = p[2]; v3 = p[3];
            }
        }
        As[r][col + 0] = v0; As[r][col + 1] = v1;
        As[r][col + 2] = v2; As[r][col + 3] = v3;
    }
    __syncthreads();

    const int lane = tid & 31;
    const int wv   = tid >> 5;       // wave id 0..3
    const int m    = lane & 15;
    const int hi   = lane >> 4;

    v8f zero = {};
    v8f acc[TPW];
#pragma unroll
    for (int i = 0; i < TPW; ++i) acc[i] = zero;

#pragma unroll
    for (int s = 0; s < 16; ++s) {
        // A fragment (matches packed B): lane<16 e0-15 -> K s*32+0..15,
        // lane>=16 e0-15 -> K s*32+16..31.
        const _Float16* ap = &As[m][s * 32 + hi * 16];
        v8h alo = *(const v8h*)ap;            // ds_load_b128
        v8h ahi = *(const v8h*)(ap + 8);      // ds_load_b128
        v16h a  = __builtin_shufflevector(alo, ahi,
                    0, 1, 2, 3, 4, 5, 6, 7, 8, 9, 10, 11, 12, 13, 14, 15);
#pragma unroll
        for (int i = 0; i < TPW; ++i) {
            int t = wv + i * 4;
            v16h b = *(const v16h*)(Wp + (((size_t)(t * 16 + s)) * 32 + lane) * 16);
            acc[i] = __builtin_amdgcn_wmma_f32_16x16x32_f16(
                         false, a, false, b, (short)0, acc[i], false, false);
        }
    }

    // ---- epilogue: bias (+ReLU), store. C layout: VGPR r -> row r + hi*8, col = lane&15.
#pragma unroll
    for (int i = 0; i < TPW; ++i) {
        int t     = wv + i * 4;
        int col   = t * 16 + (lane & 15);
        bool cOk  = col < NcolsReal;
        float bv  = cOk ? bias[col] : 0.0f;
#pragma unroll
        for (int r = 0; r < 8; ++r) {
            int row = mbase + r + hi * 8;
            if (cOk && row < M) {
                float v = acc[i][r] + bv;
                if (RELU) v = v > 0.0f ? v : 0.0f;
                if (OUTF16)
                    ((_Float16*)outp)[(size_t)row * outStride + col] = (_Float16)v;
                else
                    ((float*)outp)[(size_t)row * outStride + col] = v;
            }
        }
    }
}

// ---------------------------------------------------------------------------
extern "C" void kernel_launch(void* const* d_in, const int* in_sizes, int n_in,
                              void* d_out, int out_size, void* d_ws, size_t ws_size,
                              hipStream_t stream)
{
    (void)in_sizes; (void)n_in; (void)out_size; (void)ws_size;

    const float* feats = (const float*)d_in[0];
    const int*   src0  = (const int*)d_in[1];
    const int*   src1  = (const int*)d_in[3];
    const int*   src2  = (const int*)d_in[5];
    const float* Ws0   = (const float*)d_in[7];
    const float* Wn0   = (const float*)d_in[8];
    const float* b0    = (const float*)d_in[9];
    const float* Ws1   = (const float*)d_in[10];
    const float* Wn1   = (const float*)d_in[11];
    const float* b1    = (const float*)d_in[12];
    const float* Ws2   = (const float*)d_in[13];
    const float* Wn2   = (const float*)d_in[14];
    const float* b2    = (const float*)d_in[15];

    // workspace layout (~119 MB)
    char* ws = (char*)d_ws;
    size_t off = 0;
    _Float16* agg = (_Float16*)(ws + off); off += (size_t)N1_ * FDIM * 2;  // 56.3 MB
    _Float16* h1  = (_Float16*)(ws + off); off += (size_t)N1_ * FDIM * 2;  // 56.3 MB
    _Float16* h2  = (_Float16*)(ws + off); off += (size_t)N2_ * FDIM * 2;  //  5.6 MB
    _Float16* Wp0 = (_Float16*)(ws + off); off += (size_t)16 * 8192 * 2;
    _Float16* Wp1 = (_Float16*)(ws + off); off += (size_t)16 * 8192 * 2;
    _Float16* Wp2 = (_Float16*)(ws + off); off += (size_t)4  * 8192 * 2;

    // pack weights into WMMA B-fragment layout
    pack_w_kernel<<<(16 * 8192 + 255) / 256, 256, 0, stream>>>(Ws0, Wn0, FDIM, 16, Wp0);
    pack_w_kernel<<<(16 * 8192 + 255) / 256, 256, 0, stream>>>(Ws1, Wn1, FDIM, 16, Wp1);
    pack_w_kernel<<<( 4 * 8192 + 255) / 256, 256, 0, stream>>>(Ws2, Wn2, CLS_, 4, Wp2);

    // layer 0: agg over input_feats (f32) -> GEMM -> relu -> h1 (f16)
    sage_agg_kernel<true ><<<N1_, FDIM, 0, stream>>>(feats, src0, agg);
    sage_gemm_kernel<4, true, true, true><<<N1_ / 16, 128, 0, stream>>>(
        feats, agg, Wp0, b0, N1_, FDIM, FDIM, h1);

    // layer 1: agg over h1 (f16) -> GEMM -> relu -> h2 (f16)
    sage_agg_kernel<false><<<N2_, FDIM, 0, stream>>>(h1, src1, agg);
    sage_gemm_kernel<4, false, true, true><<<(N2_ + 15) / 16, 128, 0, stream>>>(
        h1, agg, Wp1, b1, N2_, FDIM, FDIM, h2);

    // layer 2: agg over h2 -> GEMM (no relu) -> f32 logits [1024 x 47]
    sage_agg_kernel<false><<<N3_, FDIM, 0, stream>>>(h2, src2, agg);
    sage_gemm_kernel<1, false, false, false><<<N3_ / 16, 128, 0, stream>>>(
        h2, agg, Wp2, b2, N3_, CLS_, CLS_, d_out);
}
